// MultiHeadAttention_81647328297462
// MI455X (gfx1250) — matrix operand order repaired and checked by reading, not compiled, not run on hardware
//
#include <hip/hip_runtime.h>
#include <hip/hip_bf16.h>

typedef __attribute__((ext_vector_type(16))) __bf16 bf16x16;
typedef __attribute__((ext_vector_type(8)))  __bf16 bf16x8;
typedef __attribute__((ext_vector_type(8)))  float  f32x8;

#define D_MODEL   1024
#define NUM_HEADS 16
#define DK        64
#define SEQ       2048
#define BATCH     2
#define MROWS     (BATCH*SEQ)   // 4096

static __device__ __forceinline__ bf16x16 combine8(bf16x8 lo, bf16x8 hi) {
  return __builtin_shufflevector(lo, hi, 0,1,2,3,4,5,6,7,8,9,10,11,12,13,14,15);
}

// A fragment (16x32 bf16): lane&15 = row M; elements = K[half*8..+7] and K[16+half*8..+7]
static __device__ __forceinline__ bf16x16 load_frag_a(const __bf16* rowptr, int half) {
  bf16x8 lo = *reinterpret_cast<const bf16x8*>(rowptr + half * 8);
  bf16x8 hi = *reinterpret_cast<const bf16x8*>(rowptr + 16 + half * 8);
  return combine8(lo, hi);
}
// B fragment (32x16 bf16): lane&15 = col N; elements = K[half*8..+7] and K[16+half*8..+7]
static __device__ __forceinline__ bf16x16 load_frag_b(const __bf16* colptr, int half) {
  bf16x8 lo = *reinterpret_cast<const bf16x8*>(colptr + half * 8);
  bf16x8 hi = *reinterpret_cast<const bf16x8*>(colptr + 16 + half * 8);
  return combine8(lo, hi);
}

static __device__ __forceinline__ f32x8 wmma_bf16(bf16x16 a, bf16x16 b, f32x8 c) {
  return __builtin_amdgcn_wmma_f32_16x16x32_bf16(false, a, false, b, (short)0, c,
                                                 false, false);
}

// ---------------- RoPE + convert to bf16 ----------------
__global__ void rope_to_bf16_kernel(const float* __restrict__ x,
                                    __bf16* __restrict__ y, int total) {
  int idx = blockIdx.x * blockDim.x + threadIdx.x;
  if (idx >= total) return;
  int d = idx & (D_MODEL - 1);
  int s = (idx >> 10) & (SEQ - 1);
  int base = idx - d;
  int j = d >> 1;                                   // theta index (repeat-2 interleave)
  // theta = 10000^(-2j/D) = exp(-j * ln(10000)/512)
  float theta = __expf(-(float)j * 0.0179889460390159797f);
  float ang = (float)(s + 1) * theta;               // positions start at 1
  float cs = cosf(ang), sn = sinf(ang);
  float rot = (d < D_MODEL / 2) ? -x[base + d + D_MODEL / 2]
                                :  x[base + d - D_MODEL / 2];
  y[idx] = (__bf16)(x[idx] * cs + rot * sn);
}

__global__ void cvt_to_bf16_kernel(const float* __restrict__ x,
                                   __bf16* __restrict__ y, int n) {
  int idx = blockIdx.x * blockDim.x + threadIdx.x;
  if (idx < n) y[idx] = (__bf16)x[idx];
}

// ---------------- GEMM: Y[M,N] = X[M,K] @ W[N,K]^T + bias ----------------
// Double-buffered LDS staging (k-step 32), 128x128 block tile, 8 waves.
// mode 0: fp32 out [M,N] ; mode 1: bf16 out in (B,H,S,DK) head layout
__global__ __launch_bounds__(256) void gemm_nt_bf16_kernel(
    const __bf16* __restrict__ X, const __bf16* __restrict__ W,
    const float* __restrict__ bias, float* __restrict__ outF,
    __bf16* __restrict__ outB, int mode) {
  const int K = 1024;
  const int LS = 40;                        // padded LDS row stride (32+8 bf16)
  __shared__ __bf16 Xs[2][128 * 40];
  __shared__ __bf16 Ws[2][128 * 40];
  const int t = threadIdx.x;
  const int lane = t & 31, wv = t >> 5;
  const int lane15 = lane & 15, half = lane >> 4;
  const int waveCol = wv & 1;               // 2 col groups of 64
  const int waveRow = wv >> 1;              // 4 row groups of 32
  const int rowBase = blockIdx.y * 128;
  const int colBase = blockIdx.x * 128;

  f32x8 c[2][4];
  for (int fr = 0; fr < 2; fr++)
    for (int fc = 0; fc < 4; fc++)
      c[fr][fc] = f32x8{0.f, 0.f, 0.f, 0.f, 0.f, 0.f, 0.f, 0.f};

  // stage tile 0 (512 chunks of 8 bf16 per matrix, 2 per thread)
  for (int p = 0; p < 2; p++) {
    int chunk = p * 256 + t;
    int r = chunk >> 2, co = (chunk & 3) * 8;
    *reinterpret_cast<bf16x8*>(&Xs[0][r * LS + co]) =
        *reinterpret_cast<const bf16x8*>(X + (size_t)(rowBase + r) * K + co);
    *reinterpret_cast<bf16x8*>(&Ws[0][r * LS + co]) =
        *reinterpret_cast<const bf16x8*>(W + (size_t)(colBase + r) * K + co);
  }
  __syncthreads();

  for (int i = 0; i < K / 32; i++) {
    const int cur = i & 1;
    const int kb_next = (i + 1) * 32;
    bf16x8 xreg[2], wreg[2];
    if (kb_next < K) {
      // fetch next tile into registers while WMMAs run on current tile
      for (int p = 0; p < 2; p++) {
        int chunk = p * 256 + t;
        int r = chunk >> 2, co = (chunk & 3) * 8;
        xreg[p] = *reinterpret_cast<const bf16x8*>(
            X + (size_t)(rowBase + r) * K + kb_next + co);
        wreg[p] = *reinterpret_cast<const bf16x8*>(
            W + (size_t)(colBase + r) * K + kb_next + co);
      }
      if (kb_next + 32 < K) {   // warm L2/WGP$ one tile further ahead
        __builtin_prefetch(X + (size_t)(rowBase + (t >> 2)) * K + kb_next + 32, 0, 1);
        __builtin_prefetch(W + (size_t)(colBase + (t >> 2)) * K + kb_next + 32, 0, 1);
      }
    }

    bf16x16 a[2], b[4];
    for (int fr = 0; fr < 2; fr++) {
      int r = waveRow * 32 + fr * 16 + lane15;
      a[fr] = load_frag_a(&Xs[cur][r * LS], half);
    }
    for (int fc = 0; fc < 4; fc++) {
      int n = waveCol * 64 + fc * 16 + lane15;
      b[fc] = load_frag_b(&Ws[cur][n * LS], half);
    }
    for (int fr = 0; fr < 2; fr++)
      for (int fc = 0; fc < 4; fc++)
        c[fr][fc] = wmma_bf16(a[fr], b[fc], c[fr][fc]);

    if (kb_next < K) {
      for (int p = 0; p < 2; p++) {
        int chunk = p * 256 + t;
        int r = chunk >> 2, co = (chunk & 3) * 8;
        *reinterpret_cast<bf16x8*>(&Xs[1 - cur][r * LS + co]) = xreg[p];
        *reinterpret_cast<bf16x8*>(&Ws[1 - cur][r * LS + co]) = wreg[p];
      }
    }
    __syncthreads();
  }

  for (int fr = 0; fr < 2; fr++) {
    for (int fc = 0; fc < 4; fc++) {
      int n = colBase + waveCol * 64 + fc * 16 + lane15;
      float bv = bias[n];
      for (int r = 0; r < 8; r++) {
        int m = rowBase + waveRow * 32 + fr * 16 + r + 8 * half;
        float v = c[fr][fc][r] + bv;
        if (mode == 0) {
          outF[(size_t)m * D_MODEL + n] = v;
        } else {
          int b_ = m >> 11, s = m & (SEQ - 1);
          int h = n >> 6, dk = n & 63;
          outB[(((size_t)b_ * NUM_HEADS + h) * SEQ + s) * DK + dk] = (__bf16)v;
        }
      }
    }
  }
}

// ---------------- Flash attention: per (64-query tile, batch*head) ----------------
// Double-buffered K/V tiles; 4 waves, each owns a full 16-row band.
__global__ __launch_bounds__(128) void attn_kernel(
    const __bf16* __restrict__ Q, const __bf16* __restrict__ Kmat,
    const __bf16* __restrict__ V, __bf16* __restrict__ O) {
  const int LS = 72;
  __shared__ __bf16 Ks[2][64 * 72];     // [key][dk]
  __shared__ __bf16 VTs[2][64 * 72];    // [dk][key] (transposed)
  __shared__ __bf16 Ps[4 * 16 * 72];    // per-wave P bounce buffer
  const int t = threadIdx.x;
  const int lane = t & 31, wv = t >> 5;
  const int lane15 = lane & 15, half = lane >> 4;
  const int qt = blockIdx.x;            // query tile (64 rows)
  const int bh = blockIdx.y;            // batch*head
  const __bf16* Qb = Q + (size_t)bh * SEQ * DK;
  const __bf16* Kb = Kmat + (size_t)bh * SEQ * DK;
  const __bf16* Vb = V + (size_t)bh * SEQ * DK;

  // Q fragments for this wave's 16 rows (registers for the whole pass)
  const int qrow = qt * 64 + wv * 16 + lane15;
  bf16x16 aq[2];
  for (int ks = 0; ks < 2; ks++)
    aq[ks] = load_frag_a(Qb + (size_t)qrow * DK + ks * 32, half);

  float mi[8], li[8];
  f32x8 o[4];
  for (int r = 0; r < 8; r++) { mi[r] = -3.0e38f; li[r] = 0.f; }
  for (int fc = 0; fc < 4; fc++)
    o[fc] = f32x8{0.f, 0.f, 0.f, 0.f, 0.f, 0.f, 0.f, 0.f};

  // stage tile 0: 512 chunks of 8 bf16, 4 per thread
  for (int p = 0; p < 4; p++) {
    int chunk = p * 128 + t;
    int r = chunk >> 3, co = (chunk & 7) * 8;
    *reinterpret_cast<bf16x8*>(&Ks[0][r * LS + co]) =
        *reinterpret_cast<const bf16x8*>(Kb + (size_t)r * DK + co);
    bf16x8 vvv = *reinterpret_cast<const bf16x8*>(Vb + (size_t)r * DK + co);
    for (int j = 0; j < 8; j++) VTs[0][(co + j) * LS + r] = vvv[j];
  }
  __syncthreads();

  for (int kt = 0; kt < SEQ / 64; kt++) {
    const int cur = kt & 1;
    const bool more = (kt + 1) < SEQ / 64;
    bf16x8 kreg[4], vreg[4];
    if (more) {
      for (int p = 0; p < 4; p++) {
        int chunk = p * 128 + t;
        int r = chunk >> 3, co = (chunk & 7) * 8;
        size_t g = (size_t)((kt + 1) * 64 + r) * DK + co;
        kreg[p] = *reinterpret_cast<const bf16x8*>(Kb + g);
        vreg[p] = *reinterpret_cast<const bf16x8*>(Vb + g);
      }
      if (kt + 2 < SEQ / 64) {
        __builtin_prefetch(Kb + (size_t)((kt + 2) * 64 + (t >> 1)) * DK, 0, 1);
        __builtin_prefetch(Vb + (size_t)((kt + 2) * 64 + (t >> 1)) * DK, 0, 1);
      }
    }

    // scores S = Q @ K^T (d_k = 64 -> 2 k-steps of 32)
    f32x8 s[4];
    for (int j = 0; j < 4; j++) {
      s[j] = f32x8{0.f, 0.f, 0.f, 0.f, 0.f, 0.f, 0.f, 0.f};
      for (int ks = 0; ks < 2; ks++) {
        bf16x16 bk = load_frag_b(&Ks[cur][(j * 16 + lane15) * LS + ks * 32], half);
        s[j] = wmma_bf16(aq[ks], bk, s[j]);
      }
    }

    // online softmax (rows live in 16-lane halves; shfl_xor 1/2/4/8 reduces cols)
    float mnew[8], alpha[8];
    for (int r = 0; r < 8; r++) {
      float mx = -3.0e38f;
      for (int j = 0; j < 4; j++) mx = fmaxf(mx, s[j][r] * 0.125f);
      for (int msk = 1; msk < 16; msk <<= 1) mx = fmaxf(mx, __shfl_xor(mx, msk, 32));
      mnew[r] = fmaxf(mi[r], mx);
      alpha[r] = __expf(mi[r] - mnew[r]);
    }
    float rs[8] = {0.f, 0.f, 0.f, 0.f, 0.f, 0.f, 0.f, 0.f};
    for (int j = 0; j < 4; j++) {
      for (int r = 0; r < 8; r++) {
        float p = __expf(s[j][r] * 0.125f - mnew[r]);
        rs[r] += p;
        Ps[(wv * 16 + r + 8 * half) * LS + j * 16 + lane15] = (__bf16)p;
      }
    }
    for (int r = 0; r < 8; r++) {
      float sum = rs[r];
      for (int msk = 1; msk < 16; msk <<= 1) sum += __shfl_xor(sum, msk, 32);
      li[r] = li[r] * alpha[r] + sum;
      mi[r] = mnew[r];
    }
    for (int fc = 0; fc < 4; fc++)
      for (int r = 0; r < 8; r++) o[fc][r] *= alpha[r];

    // O += P @ V  (P via per-wave LDS bounce to get A-fragment layout)
    for (int ks = 0; ks < 2; ks++) {
      bf16x16 ap = load_frag_a(&Ps[(wv * 16 + lane15) * LS + ks * 32], half);
      for (int fc = 0; fc < 4; fc++) {
        bf16x16 bv = load_frag_b(&VTs[cur][(fc * 16 + lane15) * LS + ks * 32], half);
        o[fc] = wmma_bf16(ap, bv, o[fc]);
      }
    }

    if (more) {   // commit next tile into the alternate buffers
      for (int p = 0; p < 4; p++) {
        int chunk = p * 128 + t;
        int r = chunk >> 3, co = (chunk & 7) * 8;
        *reinterpret_cast<bf16x8*>(&Ks[1 - cur][r * LS + co]) = kreg[p];
        for (int j = 0; j < 8; j++) VTs[1 - cur][(co + j) * LS + r] = vreg[p][j];
      }
    }
    __syncthreads();
  }

  // epilogue: O / l, store concat-head bf16 layout [B,S,D]
  const int b_ = bh >> 4, h = bh & 15;
  for (int r = 0; r < 8; r++) {
    float inv = 1.f / li[r];
    int srow = qt * 64 + wv * 16 + r + 8 * half;
    for (int fc = 0; fc < 4; fc++) {
      int col = h * 64 + fc * 16 + lane15;
      O[((size_t)b_ * SEQ + srow) * D_MODEL + col] = (__bf16)(o[fc][r] * inv);
    }
  }
}

extern "C" void kernel_launch(void* const* d_in, const int* in_sizes, int n_in,
                              void* d_out, int out_size, void* d_ws, size_t ws_size,
                              hipStream_t stream) {
  (void)in_sizes; (void)n_in; (void)out_size; (void)ws_size;
  const float* q    = (const float*)d_in[0];
  const float* k    = (const float*)d_in[1];
  const float* v    = (const float*)d_in[2];
  const float* Wq_w = (const float*)d_in[3];
  const float* Wq_b = (const float*)d_in[4];
  const float* Wk_w = (const float*)d_in[5];
  const float* Wk_b = (const float*)d_in[6];
  const float* Wv_w = (const float*)d_in[7];
  const float* Wv_b = (const float*)d_in[8];
  const float* Wo_w = (const float*)d_in[9];
  const float* Wo_b = (const float*)d_in[10];

  const size_t ACT = (size_t)BATCH * SEQ * D_MODEL;    // 4,194,304 elems
  const size_t WEL = (size_t)D_MODEL * D_MODEL;        // 1,048,576 elems
  char* ws = (char*)d_ws;
  size_t off = 0;
  auto carve = [&](size_t elems) { __bf16* p = (__bf16*)(ws + off); off += elems * 2; return p; };
  __bf16* qr = carve(ACT);
  __bf16* kr = carve(ACT);
  __bf16* vb = carve(ACT);
  __bf16* wq = carve(WEL);
  __bf16* wk = carve(WEL);
  __bf16* wv = carve(WEL);
  __bf16* wo = carve(WEL);
  __bf16* Qh = carve(ACT);
  __bf16* Kh = carve(ACT);
  __bf16* Vh = carve(ACT);
  __bf16* Ob = carve(ACT);

  const int cvtBlocks = (int)(WEL / 256);
  cvt_to_bf16_kernel<<<cvtBlocks, 256, 0, stream>>>(Wq_w, wq, (int)WEL);
  cvt_to_bf16_kernel<<<cvtBlocks, 256, 0, stream>>>(Wk_w, wk, (int)WEL);
  cvt_to_bf16_kernel<<<cvtBlocks, 256, 0, stream>>>(Wv_w, wv, (int)WEL);
  cvt_to_bf16_kernel<<<cvtBlocks, 256, 0, stream>>>(Wo_w, wo, (int)WEL);

  const int actBlocks = (int)(ACT / 256);
  rope_to_bf16_kernel<<<actBlocks, 256, 0, stream>>>(q, qr, (int)ACT);
  rope_to_bf16_kernel<<<actBlocks, 256, 0, stream>>>(k, kr, (int)ACT);
  cvt_to_bf16_kernel<<<actBlocks, 256, 0, stream>>>(v, vb, (int)ACT);

  dim3 gemmGrid(D_MODEL / 128, MROWS / 128);   // (8, 32)
  gemm_nt_bf16_kernel<<<gemmGrid, 256, 0, stream>>>(qr, wq, Wq_b, nullptr, Qh, 1);
  gemm_nt_bf16_kernel<<<gemmGrid, 256, 0, stream>>>(kr, wk, Wk_b, nullptr, Kh, 1);
  gemm_nt_bf16_kernel<<<gemmGrid, 256, 0, stream>>>(vb, wv, Wv_b, nullptr, Vh, 1);

  attn_kernel<<<dim3(SEQ / 64, BATCH * NUM_HEADS), 128, 0, stream>>>(Qh, Kh, Vh, Ob);

  gemm_nt_bf16_kernel<<<gemmGrid, 256, 0, stream>>>(Ob, wo, Wo_b, (float*)d_out,
                                                    nullptr, 0);
}
